// DecoderBlock_21165598835361
// MI455X (gfx1250) — matrix operand order repaired and checked by reading, compiled, not run on hardware
//
#include <hip/hip_runtime.h>
#include <hip/hip_bf16.h>
#include <math.h>

typedef _Float16 half_t;
typedef __attribute__((ext_vector_type(16))) _Float16 v16h;
typedef __attribute__((ext_vector_type(8)))  _Float16 v8h;
typedef __attribute__((ext_vector_type(8)))  float    v8f;
typedef __attribute__((ext_vector_type(4)))  _Float16 v4h;

#define D_MODEL 1024
#define D3      3072
#define D_MID   4096
#define NHEAD   16
#define HDIM    64
#define BATCH   2
#define SEQ     2048
#define ROWS    (BATCH*SEQ)

// ---------------- async global->LDS (CDNA5, ASYNCcnt) -----------------------

#if __has_builtin(__builtin_amdgcn_global_load_async_to_lds_b128) && \
    __has_builtin(__builtin_amdgcn_s_wait_asynccnt)
#define USE_ASYNC_LDS 1
#endif

typedef int vi4 __attribute__((vector_size(16)));
typedef __attribute__((address_space(1))) vi4 GVI4;   // global int4
typedef __attribute__((address_space(3))) vi4 LVI4;   // LDS int4

__device__ __forceinline__ void async_b128(const half_t* g, half_t* l) {
#ifdef USE_ASYNC_LDS
  __builtin_amdgcn_global_load_async_to_lds_b128((GVI4*)g, (LVI4*)l, 0, 0);
#else
  *(v8h*)l = *(const v8h*)g;
#endif
}

__device__ __forceinline__ void wait_async0() {
#ifdef USE_ASYNC_LDS
  __builtin_amdgcn_s_wait_asynccnt(0);
#endif
}

// ---------------- WMMA fragment helpers (CDNA5 wave32 layouts, ISA 7.12.2) ---

// A 16x32 f16: lanes 0-15 row=lane, halves 0..7 = K0..7, 8..15 = K16..23
//              lanes 16-31 row=lane-16, halves 0..7 = K8..15, 8..15 = K24..31
__device__ __forceinline__ v16h load_a_frag(const half_t* p0, int ld) {
  const int lane = threadIdx.x & 31;
  const int row  = lane & 15;
  const int koff = (lane >> 4) << 3;   // 0 or 8
  const half_t* p = p0 + row * ld + koff;
  v16h a;
#pragma unroll
  for (int i = 0; i < 8; ++i) { a[i] = p[i]; a[8 + i] = p[16 + i]; }
  return a;
}

// B 32x16 f16 from K-contiguous (transposed) storage [n][k]:
// lanes 0-15: n=lane, halves 0..7 = K0..7, 8..15 = K16..23 ; lanes 16-31: n=lane-16, K8..15, K24..31
__device__ __forceinline__ v16h load_b_frag(const half_t* p0, int ld) {
  const int lane = threadIdx.x & 31;
  const int n    = lane & 15;
  const int koff = (lane >> 4) << 3;   // 0 or 8
  const half_t* p = p0 + n * ld + koff;
  v16h b;
#pragma unroll
  for (int i = 0; i < 8; ++i) { b[i] = p[i]; b[8 + i] = p[16 + i]; }
  return b;
}

__device__ __forceinline__ v8f wmma_f16(v16h a, v16h b, v8f c) {
  return __builtin_amdgcn_wmma_f32_16x16x32_f16(false, a, false, b, (short)0, c,
                                                false, false);
}

// ---------------- elementwise kernels ---------------------------------------

__global__ __launch_bounds__(256) void cvt_f16_kernel(const float* __restrict__ in,
                                                      half_t* __restrict__ out) {
  const size_t i = ((size_t)blockIdx.x * 256 + threadIdx.x) * 4;
  const float4 v = *(const float4*)(in + i);
  v4h o; o[0] = (half_t)v.x; o[1] = (half_t)v.y; o[2] = (half_t)v.z; o[3] = (half_t)v.w;
  *(v4h*)(out + i) = o;
}

__global__ __launch_bounds__(256) void ln_f16_kernel(const float* __restrict__ x,
                                                     const float* __restrict__ g,
                                                     const float* __restrict__ be,
                                                     half_t* __restrict__ out) {
  const int row = blockIdx.x;
  const int t = threadIdx.x;
  const float4 v = ((const float4*)(x + (size_t)row * D_MODEL))[t];
  float s  = v.x + v.y + v.z + v.w;
  float s2 = v.x * v.x + v.y * v.y + v.z * v.z + v.w * v.w;
#pragma unroll
  for (int m = 1; m < 32; m <<= 1) {
    s  += __shfl_xor(s, m, 32);
    s2 += __shfl_xor(s2, m, 32);
  }
  __shared__ float rs_[8], rs2_[8];
  if ((t & 31) == 0) { rs_[t >> 5] = s; rs2_[t >> 5] = s2; }
  __syncthreads();
  float tot = 0.0f, tot2 = 0.0f;
#pragma unroll
  for (int i = 0; i < 8; ++i) { tot += rs_[i]; tot2 += rs2_[i]; }
  const float mean = tot * (1.0f / D_MODEL);
  const float var  = tot2 * (1.0f / D_MODEL) - mean * mean;
  const float rstd = rsqrtf(var + 1e-5f);
  const float4 gv = ((const float4*)g)[t];
  const float4 bv = ((const float4*)be)[t];
  v4h o;
  o[0] = (half_t)((v.x - mean) * rstd * gv.x + bv.x);
  o[1] = (half_t)((v.y - mean) * rstd * gv.y + bv.y);
  o[2] = (half_t)((v.z - mean) * rstd * gv.z + bv.z);
  o[3] = (half_t)((v.w - mean) * rstd * gv.w + bv.w);
  ((v4h*)(out + (size_t)row * D_MODEL))[t] = o;
}

// ---------------- tiled WMMA GEMM with fused epilogues ----------------------
// Block tile 128x64, 8 waves in 4x2 grid, each wave owns 32x32 (4 C frags).
// A tile: async global->LDS (ASYNCcnt), double-buffered; B tile: regs +
// transpose scatter, double-buffered. One barrier per K-step.

enum { EPI_H = 0, EPI_GELU_H = 1, EPI_RES_F = 2 };

__device__ __forceinline__ float gelu_f(float x) {
  const float c = 0.7978845608028654f;
  return 0.5f * x * (1.0f + tanhf(c * (x + 0.044715f * x * x * x)));
}

template <int EPI>
__global__ __launch_bounds__(256) void gemm_wmma_kernel(
    const half_t* __restrict__ A, const half_t* __restrict__ Bw,
    const float* __restrict__ bias, const float* __restrict__ resid,
    float* __restrict__ outF, half_t* __restrict__ outH, int M, int N, int K) {
  (void)M;
  __shared__ half_t As[2][128][40];  // [buf][m][k]
  __shared__ half_t Bs[2][64][40];   // [buf][n][k] (transposed)
  const int t   = threadIdx.x;
  const int bm  = blockIdx.y << 7;
  const int bn  = blockIdx.x << 6;
  const int wid = t >> 5;
  const int wm  = (wid & 3) << 5;   // wave row offset 0,32,64,96
  const int wn  = (wid >> 2) << 5;  // wave col offset 0,32
  v8f acc00 = {}, acc01 = {}, acc10 = {}, acc11 = {};
  const int arow = t >> 1, acol = (t & 1) << 4;   // A tile: 128 rows x 32 k
  const int brow = t >> 3, bcol = (t & 7) << 3;   // B tile: 32 k x 64 n

  const half_t* ap = A  + (size_t)(bm + arow) * K + acol;
  const half_t* bp = Bw + (size_t)brow * N + bn + bcol;
  async_b128(ap,     &As[0][arow][acol]);
  async_b128(ap + 8, &As[0][arow][acol + 8]);
  v4h b0 = *(const v4h*)bp;
  v4h b1 = *(const v4h*)(bp + 4);

  for (int kk = 0; kk < K; kk += 32) {
    const int cur = (kk >> 5) & 1;
#pragma unroll
    for (int i = 0; i < 4; ++i) Bs[cur][bcol + i][brow]     = b0[i];
#pragma unroll
    for (int i = 0; i < 4; ++i) Bs[cur][bcol + 4 + i][brow] = b1[i];
    wait_async0();            // this tile's async A copies landed
    __syncthreads();          // publish tile 'cur' block-wide
    if (kk + 32 < K) {        // safe post-barrier: everyone left buffer cur^1
      const half_t* apn = ap + kk + 32;
      const half_t* bpn = bp + (size_t)(kk + 32) * N;
      async_b128(apn,     &As[cur ^ 1][arow][acol]);
      async_b128(apn + 8, &As[cur ^ 1][arow][acol + 8]);
      b0 = *(const v4h*)bpn;
      b1 = *(const v4h*)(bpn + 4);
      if (kk + 64 < K) __builtin_prefetch(bp + (size_t)(kk + 64) * N, 0, 0);
    }
    v16h af0 = load_a_frag(&As[cur][wm][0], 40);
    v16h af1 = load_a_frag(&As[cur][wm + 16][0], 40);
    v16h bf0 = load_b_frag(&Bs[cur][wn][0], 40);
    v16h bf1 = load_b_frag(&Bs[cur][wn + 16][0], 40);
    acc00 = wmma_f16(af0, bf0, acc00);
    acc01 = wmma_f16(af0, bf1, acc01);
    acc10 = wmma_f16(af1, bf0, acc10);
    acc11 = wmma_f16(af1, bf1, acc11);
  }

  const int lane  = t & 31;
  const int cn    = lane & 15;
  const int rbase = (lane >> 4) << 3;
#pragma unroll
  for (int fr = 0; fr < 2; ++fr) {
#pragma unroll
    for (int fc = 0; fc < 2; ++fc) {
      v8f acc = fr ? (fc ? acc11 : acc10) : (fc ? acc01 : acc00);
      const int gcol = bn + wn + fc * 16 + cn;
      const float bv = bias[gcol];
#pragma unroll
      for (int v = 0; v < 8; ++v) {
        const int grow = bm + wm + fr * 16 + rbase + v;
        float val = acc[v] + bv;
        if (EPI == EPI_GELU_H) val = gelu_f(val);
        if (EPI == EPI_RES_F) {
          outF[(size_t)grow * N + gcol] = val + resid[(size_t)grow * N + gcol];
        } else {
          outH[(size_t)grow * N + gcol] = (half_t)val;
        }
      }
    }
  }
}

// ---------------- flash attention (causal, scale 1/sqrt(D_MODEL)) -----------
// K tile async global->LDS double-buffered; V tile regs + transpose scatter,
// double-buffered. One barrier per key-block.

__global__ __launch_bounds__(256) void attn_kernel(const half_t* __restrict__ qkv,
                                                   half_t* __restrict__ ctx) {
  __shared__ half_t Ks[2][32][72];   // [buf][key][hd]
  __shared__ half_t Vt[2][64][40];   // [buf][hd][key] (transposed)
  __shared__ half_t Ps[8][16][40];   // per-wave P tile [qrow][key]
  const int t    = threadIdx.x;
  const int wid  = t >> 5;
  const int lane = t & 31;
  const int b = blockIdx.y >> 4;
  const int h = blockIdx.y & 15;
  const int q0 = (blockIdx.x << 7) + (wid << 4);   // this wave's first query

  const half_t* qp = qkv + (size_t)(b * SEQ + q0) * D3 + h * HDIM;
  v16h qf0 = load_a_frag(qp, D3);
  v16h qf1 = load_a_frag(qp + 32, D3);

  v8f o0 = {}, o1 = {}, o2 = {}, o3 = {};
  float mrow[8], lrow[8];
#pragma unroll
  for (int v = 0; v < 8; ++v) { mrow[v] = -1e30f; lrow[v] = 0.0f; }

  const int cn    = lane & 15;
  const int rbase = (lane >> 4) << 3;
  const int kkey  = t >> 3;           // K/V tile loader: 32 keys x 64 hd
  const int kcol  = (t & 7) << 3;
  const int qlast = (blockIdx.x << 7) + 127;

  const half_t* kvbase = qkv + (size_t)(b * SEQ + kkey) * D3 + D_MODEL + h * HDIM + kcol;
  async_b128(kvbase, &Ks[0][kkey][kcol]);           // K tile kb=0
  v4h v0 = *(const v4h*)(kvbase + D_MODEL);         // V tile kb=0
  v4h v1 = *(const v4h*)(kvbase + D_MODEL + 4);

  for (int kb = 0; kb <= qlast; kb += 32) {
    const int cur = (kb >> 5) & 1;
#pragma unroll
    for (int i = 0; i < 4; ++i) Vt[cur][kcol + i][kkey]     = v0[i];
#pragma unroll
    for (int i = 0; i < 4; ++i) Vt[cur][kcol + 4 + i][kkey] = v1[i];
    wait_async0();
    __syncthreads();
    if (kb + 32 <= qlast) {   // issue next K/V tile post-barrier
      const half_t* kp = kvbase + (size_t)(kb + 32) * D3;
      async_b128(kp, &Ks[cur ^ 1][kkey][kcol]);
      v0 = *(const v4h*)(kp + D_MODEL);
      v1 = *(const v4h*)(kp + D_MODEL + 4);
    }

    if (kb <= q0) {   // wave-uniform; q tiles 16-aligned, kb 32-aligned
      v8f s0 = {}, s1 = {};
      s0 = wmma_f16(qf0, load_b_frag(&Ks[cur][0][0], 72), s0);
      s1 = wmma_f16(qf0, load_b_frag(&Ks[cur][16][0], 72), s1);
      s0 = wmma_f16(qf1, load_b_frag(&Ks[cur][0][32], 72), s0);
      s1 = wmma_f16(qf1, load_b_frag(&Ks[cur][16][32], 72), s1);

      const float scale = 0.03125f;   // 1/sqrt(1024)
      float pm0[8], pm1[8];
#pragma unroll
      for (int v = 0; v < 8; ++v) {
        const int qrow = q0 + rbase + v;
        pm0[v] = (kb + cn      <= qrow) ? s0[v] * scale : -1e30f;
        pm1[v] = (kb + 16 + cn <= qrow) ? s1[v] * scale : -1e30f;
      }
#pragma unroll
      for (int v = 0; v < 8; ++v) {
        float mx = fmaxf(pm0[v], pm1[v]);
#pragma unroll
        for (int m = 1; m < 16; m <<= 1) mx = fmaxf(mx, __shfl_xor(mx, m, 16));
        const float mnew  = fmaxf(mrow[v], mx);
        const float alpha = __expf(mrow[v] - mnew);
        const float p0 = __expf(pm0[v] - mnew);
        const float p1 = __expf(pm1[v] - mnew);
        float psum = p0 + p1;
#pragma unroll
        for (int m = 1; m < 16; m <<= 1) psum += __shfl_xor(psum, m, 16);
        lrow[v] = lrow[v] * alpha + psum;
        mrow[v] = mnew;
        o0[v] *= alpha; o1[v] *= alpha; o2[v] *= alpha; o3[v] *= alpha;
        Ps[wid][rbase + v][cn]      = (half_t)p0;
        Ps[wid][rbase + v][16 + cn] = (half_t)p1;
      }
      asm volatile("s_wait_dscnt 0" ::: "memory");   // cross-lane LDS transpose
      v16h pf = load_a_frag(&Ps[wid][0][0], 40);
      o0 = wmma_f16(pf, load_b_frag(&Vt[cur][0][0],  40), o0);
      o1 = wmma_f16(pf, load_b_frag(&Vt[cur][16][0], 40), o1);
      o2 = wmma_f16(pf, load_b_frag(&Vt[cur][32][0], 40), o2);
      o3 = wmma_f16(pf, load_b_frag(&Vt[cur][48][0], 40), o3);
    }
  }

  half_t* cp = ctx + (size_t)(b * SEQ + q0) * D_MODEL + h * HDIM;
#pragma unroll
  for (int v = 0; v < 8; ++v) {
    const float inv = 1.0f / lrow[v];
    const size_t r = (size_t)(rbase + v) * D_MODEL;
    cp[r + cn]      = (half_t)(o0[v] * inv);
    cp[r + 16 + cn] = (half_t)(o1[v] * inv);
    cp[r + 32 + cn] = (half_t)(o2[v] * inv);
    cp[r + 48 + cn] = (half_t)(o3[v] * inv);
  }
}

// ---------------- launch ----------------------------------------------------

extern "C" void kernel_launch(void* const* d_in, const int* in_sizes, int n_in,
                              void* d_out, int out_size, void* d_ws, size_t ws_size,
                              hipStream_t stream) {
  (void)in_sizes; (void)n_in; (void)out_size; (void)ws_size;
  const float* x     = (const float*)d_in[0];
  const float* w_qkv = (const float*)d_in[1];
  const float* b_qkv = (const float*)d_in[2];
  const float* w_o   = (const float*)d_in[3];
  const float* b_o   = (const float*)d_in[4];
  const float* g1    = (const float*)d_in[5];
  const float* be1   = (const float*)d_in[6];
  const float* g2    = (const float*)d_in[7];
  const float* be2   = (const float*)d_in[8];
  const float* w1    = (const float*)d_in[9];
  const float* b1    = (const float*)d_in[10];
  const float* w2    = (const float*)d_in[11];
  const float* b2    = (const float*)d_in[12];
  float* out = (float*)d_out;

  char* ws = (char*)d_ws;
  size_t off = 0;
  auto alloc = [&](size_t bytes) {
    void* p = ws + off;
    off += (bytes + 255) & ~(size_t)255;
    return p;
  };
  half_t* nx1    = (half_t*)alloc((size_t)ROWS * D_MODEL * 2);     // ln1(x), reused as nx2
  half_t* wqkv16 = (half_t*)alloc((size_t)D_MODEL * D3 * 2);
  char*   uni    = (char*)alloc((size_t)ROWS * D3 * 2 + 256 + (size_t)ROWS * D_MODEL * 2);
  half_t* qkv16  = (half_t*)uni;
  half_t* ctx16  = (half_t*)(uni + (((size_t)ROWS * D3 * 2 + 255) & ~(size_t)255));
  half_t* h16    = (half_t*)uni;   // aliases qkv16+ctx16 (both dead before FFN1)
  half_t* wo16   = (half_t*)alloc((size_t)D_MODEL * D_MODEL * 2);
  float*  x1     = (float*)alloc((size_t)ROWS * D_MODEL * 4);
  half_t* nx2    = nx1;
  half_t* w116   = (half_t*)alloc((size_t)D_MODEL * D_MID * 2);
  half_t* w216   = (half_t*)alloc((size_t)D_MID * D_MODEL * 2);

  dim3 blk(256);
  cvt_f16_kernel<<<dim3((D_MODEL * D3) / 1024), blk, 0, stream>>>(w_qkv, wqkv16);
  cvt_f16_kernel<<<dim3((D_MODEL * D_MODEL) / 1024), blk, 0, stream>>>(w_o, wo16);
  cvt_f16_kernel<<<dim3((D_MODEL * D_MID) / 1024), blk, 0, stream>>>(w1, w116);
  cvt_f16_kernel<<<dim3((D_MID * D_MODEL) / 1024), blk, 0, stream>>>(w2, w216);

  ln_f16_kernel<<<dim3(ROWS), blk, 0, stream>>>(x, g1, be1, nx1);

  gemm_wmma_kernel<EPI_H><<<dim3(D3 / 64, ROWS / 128), blk, 0, stream>>>(
      nx1, wqkv16, b_qkv, nullptr, nullptr, qkv16, ROWS, D3, D_MODEL);

  attn_kernel<<<dim3(SEQ / 128, BATCH * NHEAD), blk, 0, stream>>>(qkv16, ctx16);

  gemm_wmma_kernel<EPI_RES_F><<<dim3(D_MODEL / 64, ROWS / 128), blk, 0, stream>>>(
      ctx16, wo16, b_o, x, x1, nullptr, ROWS, D_MODEL, D_MODEL);

  ln_f16_kernel<<<dim3(ROWS), blk, 0, stream>>>(x1, g2, be2, nx2);

  gemm_wmma_kernel<EPI_GELU_H><<<dim3(D_MID / 64, ROWS / 128), blk, 0, stream>>>(
      nx2, w116, b1, nullptr, nullptr, h16, ROWS, D_MID, D_MODEL);

  gemm_wmma_kernel<EPI_RES_F><<<dim3(D_MODEL / 64, ROWS / 128), blk, 0, stream>>>(
      h16, w216, b2, x1, out, nullptr, ROWS, D_MODEL, D_MID);
}
